// ContrastiveLoss_15762529976387
// MI455X (gfx1250) — matrix operand order repaired and checked by reading, not compiled, not run on hardware
//
#include <hip/hip_runtime.h>
#include <hip/hip_bf16.h>

// ---------------------------------------------------------------------------
// Problem: N=8192, D=128, T=0.5
// loss = -(1/(T*N)) * (sum_i z_i) . (sum_j zp_j) + sum_j log(den_sum[j])
// den_sum[j] = sum_{i != j} exp( (zn_i . zn_j) / T ),  zn = x / max(|x|,1e-8)
// z uses eps 1e-12.  Heavy part: 8192x8192x128 GEMM fused with exp + row-sum,
// done with v_wmma_f32_16x16x32_bf16 using a bf16 hi/lo split (3 WMMAs/k-step)
// for near-fp32 accuracy.
// ---------------------------------------------------------------------------

#define NROWS 8192
#define DDIM  128
#define PADK  40   // 32 K-chunk + 8 pad (80B stride: 16B aligned, bank-conflict free)

typedef __attribute__((ext_vector_type(16))) __bf16 v16bf;
typedef __attribute__((ext_vector_type(8)))  float  v8f;

union FragB {
    v16bf v;
    uint4 q[2];
};

__device__ __forceinline__ unsigned short f2bf_rne(float x) {
    unsigned u = __float_as_uint(x);
    unsigned r = u + 0x7FFFu + ((u >> 16) & 1u);
    return (unsigned short)(r >> 16);
}
__device__ __forceinline__ float bf2f(unsigned short h) {
    return __uint_as_float(((unsigned)h) << 16);
}

__device__ __forceinline__ v8f wmma_bf16(v16bf a, v16bf b, v8f c) {
    // (neg_a, A, neg_b, B, c_mod, C, reuse_a, reuse_b)
    return __builtin_amdgcn_wmma_f32_16x16x32_bf16(false, a, false, b, (short)0, c,
                                                   false, false);
}

// ---------------------------------------------------------------------------
// Kernel 1: per-row L2 norm; store norms; store bf16 hi/lo of zn (eps=1e-8).
// One wave (32 lanes * float4) per 128-float row. Rows >= NROWS -> pair matrix
// (norm only; pair rows only enter via column sums).
// ---------------------------------------------------------------------------
__global__ __launch_bounds__(256) void normalize_split(
    const float* __restrict__ nodes, const float* __restrict__ pair,
    unsigned short* __restrict__ zhi, unsigned short* __restrict__ zlo,
    float* __restrict__ norm_n, float* __restrict__ norm_p) {
    int warp = threadIdx.x >> 5;
    int lane = threadIdx.x & 31;
    int row  = blockIdx.x * 8 + warp;           // 0..16383
    bool isPair = row >= NROWS;
    int r = isPair ? row - NROWS : row;
    const float4* src =
        (const float4*)((isPair ? pair : nodes) + (size_t)r * DDIM);
    float4 v = src[lane];
    float ss = v.x * v.x + v.y * v.y + v.z * v.z + v.w * v.w;
#pragma unroll
    for (int m = 16; m >= 1; m >>= 1) ss += __shfl_xor(ss, m, 32);
    float nrm = sqrtf(ss);
    if (isPair) {
        if (lane == 0) norm_p[r] = nrm;
        return;
    }
    if (lane == 0) norm_n[r] = nrm;
    float s8 = 1.0f / fmaxf(nrm, 1e-8f);
    float xs[4] = {v.x * s8, v.y * s8, v.z * s8, v.w * s8};
    unsigned short hi[4], lo[4];
#pragma unroll
    for (int k = 0; k < 4; ++k) {
        unsigned short hb = f2bf_rne(xs[k]);
        hi[k] = hb;
        lo[k] = f2bf_rne(xs[k] - bf2f(hb));
    }
    size_t base = (size_t)r * DDIM + lane * 4;
    uint2 ph, pl;
    ph.x = (unsigned)hi[0] | ((unsigned)hi[1] << 16);
    ph.y = (unsigned)hi[2] | ((unsigned)hi[3] << 16);
    pl.x = (unsigned)lo[0] | ((unsigned)lo[1] << 16);
    pl.y = (unsigned)lo[2] | ((unsigned)lo[3] << 16);
    *(uint2*)&zhi[base] = ph;
    *(uint2*)&zlo[base] = pl;
}

// ---------------------------------------------------------------------------
// Kernel 2: chunked column sums of z (eps=1e-12). blockIdx.y: 0=nodes,1=pair.
// colpart[(which*64 + chunk)*128 + d]; fixed-order inner loop -> deterministic.
// ---------------------------------------------------------------------------
__global__ __launch_bounds__(128) void colsum_part(
    const float* __restrict__ nodes, const float* __restrict__ pair,
    const float* __restrict__ norm_n, const float* __restrict__ norm_p,
    float* __restrict__ colpart) {
    int d = threadIdx.x;      // 0..127
    int c = blockIdx.x;       // 0..63
    int which = blockIdx.y;   // 0 nodes, 1 pair
    const float* x   = which ? pair   : nodes;
    const float* nrm = which ? norm_p : norm_n;
    float acc = 0.0f;
    int i0 = c * 128;
    for (int i = i0; i < i0 + 128; ++i)
        acc += x[(size_t)i * DDIM + d] / fmaxf(nrm[i], 1e-12f);
    colpart[((size_t)which * 64 + c) * 128 + d] = acc;
}

// ---------------------------------------------------------------------------
// Kernel 3: S = zn * zn^T in 128x128 tiles, fused exp(2*S), diagonal masked,
// row-sum per tile -> partial[bn][row]. bf16 hi/lo => 3 WMMAs per k-step.
// 8 waves: wave (wm,wn) owns rows wm*32..+32, cols wn*64..+64 (2x4 WMMA tiles).
// ---------------------------------------------------------------------------
__global__ __launch_bounds__(256) void sim_exp_rowsum(
    const unsigned short* __restrict__ zhi,
    const unsigned short* __restrict__ zlo,
    float* __restrict__ partial) {
    __shared__ unsigned short Ah[128][PADK];
    __shared__ unsigned short Al[128][PADK];
    __shared__ unsigned short Bh[128][PADK];
    __shared__ unsigned short Bl[128][PADK];
    __shared__ float rowsum[128];

    int bm = blockIdx.y, bn = blockIdx.x;
    int t = threadIdx.x;
    int wave = t >> 5, lane = t & 31;
    int wm = wave >> 1, wn = wave & 1;
    int h = lane >> 4, l15 = lane & 15;

    v8f acc[2][4];
#pragma unroll
    for (int tm = 0; tm < 2; ++tm)
#pragma unroll
        for (int tn = 0; tn < 4; ++tn)
#pragma unroll
            for (int e = 0; e < 8; ++e) acc[tm][tn][e] = 0.0f;

    if (t < 128) rowsum[t] = 0.0f;

    size_t rowA = (size_t)bm * 128 * DDIM;
    size_t rowB = (size_t)bn * 128 * DDIM;

    for (int c = 0; c < 4; ++c) {   // K chunks of 32
        __syncthreads();
        // stage 128 rows x 32 K bf16 per array (4 uint4 per row)
        for (int idx = t; idx < 512; idx += 256) {
            int row = idx >> 2, seg = idx & 3;
            size_t offA = rowA + (size_t)row * DDIM + c * 32 + seg * 8;
            size_t offB = rowB + (size_t)row * DDIM + c * 32 + seg * 8;
            *(uint4*)&Ah[row][seg * 8] = *(const uint4*)(zhi + offA);
            *(uint4*)&Al[row][seg * 8] = *(const uint4*)(zlo + offA);
            *(uint4*)&Bh[row][seg * 8] = *(const uint4*)(zhi + offB);
            *(uint4*)&Bl[row][seg * 8] = *(const uint4*)(zlo + offB);
        }
        __syncthreads();

        FragB ah[2], al[2], bh[4], bl[4];
#pragma unroll
        for (int tm = 0; tm < 2; ++tm) {
            int m = wm * 32 + tm * 16 + l15;
            // A 16x32 bf16: lane half h holds K = {8h..8h+7, 16+8h..16+8h+7}
            ah[tm].q[0] = *(const uint4*)&Ah[m][8 * h];
            ah[tm].q[1] = *(const uint4*)&Ah[m][16 + 8 * h];
            al[tm].q[0] = *(const uint4*)&Al[m][8 * h];
            al[tm].q[1] = *(const uint4*)&Al[m][16 + 8 * h];
        }
#pragma unroll
        for (int tn = 0; tn < 4; ++tn) {
            int n = wn * 64 + tn * 16 + l15;
            // B 32x16 bf16: lane half h holds K = {8h..8h+7, 16+8h..16+8h+7} of column n
            bh[tn].q[0] = *(const uint4*)&Bh[n][8 * h];
            bh[tn].q[1] = *(const uint4*)&Bh[n][16 + 8 * h];
            bl[tn].q[0] = *(const uint4*)&Bl[n][8 * h];
            bl[tn].q[1] = *(const uint4*)&Bl[n][16 + 8 * h];
        }
#pragma unroll
        for (int tm = 0; tm < 2; ++tm)
#pragma unroll
            for (int tn = 0; tn < 4; ++tn) {
                acc[tm][tn] = wmma_bf16(ah[tm].v, bh[tn].v, acc[tm][tn]);
                acc[tm][tn] = wmma_bf16(ah[tm].v, bl[tn].v, acc[tm][tn]);
                acc[tm][tn] = wmma_bf16(al[tm].v, bh[tn].v, acc[tm][tn]);
            }
    }

    // epilogue: exp(2s), mask diagonal, reduce rows
    float rp[2][8];
#pragma unroll
    for (int tm = 0; tm < 2; ++tm)
#pragma unroll
        for (int r = 0; r < 8; ++r) rp[tm][r] = 0.0f;

#pragma unroll
    for (int tm = 0; tm < 2; ++tm)
#pragma unroll
        for (int tn = 0; tn < 4; ++tn) {
            int gj = bn * 128 + wn * 64 + tn * 16 + l15;
#pragma unroll
            for (int r = 0; r < 8; ++r) {
                int gi = bm * 128 + wm * 32 + tm * 16 + h * 8 + r;
                float s = acc[tm][tn][r];
                float e = (gi == gj) ? 0.0f : __expf(2.0f * s);  // 1/T = 2
                rp[tm][r] += e;
            }
        }
#pragma unroll
    for (int tm = 0; tm < 2; ++tm)
#pragma unroll
        for (int r = 0; r < 8; ++r) {
#pragma unroll
            for (int m = 1; m <= 8; m <<= 1)
                rp[tm][r] += __shfl_xor(rp[tm][r], m, 32);
        }
    if (l15 == 0) {
#pragma unroll
        for (int tm = 0; tm < 2; ++tm)
#pragma unroll
            for (int r = 0; r < 8; ++r)
                atomicAdd(&rowsum[wm * 32 + tm * 16 + h * 8 + r], rp[tm][r]);
        // exactly 2 commutative f32 adds per slot (wn=0,1) -> deterministic
    }
    __syncthreads();
    if (t < 128) partial[(size_t)bn * NROWS + bm * 128 + t] = rowsum[t];
}

// ---------------------------------------------------------------------------
// Kernel 4: den_sum[i] = sum over 64 bn-partials (fixed order).
// ---------------------------------------------------------------------------
__global__ __launch_bounds__(256) void den_reduce(const float* __restrict__ partial,
                                                  float* __restrict__ den) {
    int i = blockIdx.x * 256 + threadIdx.x;
    float s = 0.0f;
    for (int bn = 0; bn < 64; ++bn) s += partial[(size_t)bn * NROWS + i];
    den[i] = s;
}

// ---------------------------------------------------------------------------
// Kernel 5: loss = sum_j log(den[j]) - (2/N) * (sum_z . sum_zp)
// ---------------------------------------------------------------------------
__global__ __launch_bounds__(256) void finalize(const float* __restrict__ colpart,
                                                const float* __restrict__ den,
                                                float* __restrict__ out) {
    __shared__ float red[256];
    int t = threadIdx.x;
    float ls = 0.0f;
    for (int j = t; j < NROWS; j += 256) ls += __logf(den[j]);
    red[t] = ls;
    __syncthreads();
    for (int s = 128; s > 0; s >>= 1) {
        if (t < s) red[t] += red[t + s];
        __syncthreads();
    }
    float total_log = red[0];
    __syncthreads();
    float d = 0.0f;
    if (t < 128) {
        float sz = 0.0f, sp = 0.0f;
        for (int c = 0; c < 64; ++c) {
            sz += colpart[(size_t)c * 128 + t];
            sp += colpart[(size_t)(64 + c) * 128 + t];
        }
        d = sz * sp;
    }
    red[t] = d;
    __syncthreads();
    for (int s = 128; s > 0; s >>= 1) {
        if (t < s) red[t] += red[t + s];
        __syncthreads();
    }
    if (t == 0) out[0] = total_log - red[0] * (2.0f / (float)NROWS);
}

// ---------------------------------------------------------------------------
extern "C" void kernel_launch(void* const* d_in, const int* in_sizes, int n_in,
                              void* d_out, int out_size, void* d_ws, size_t ws_size,
                              hipStream_t stream) {
    const float* nodes = (const float*)d_in[0];
    const float* pair  = (const float*)d_in[1];
    // d_in[2] (labels) and d_in[3] (mask == eye) are dead inputs.
    float* out = (float*)d_out;

    char* ws = (char*)d_ws;
    unsigned short* zhi = (unsigned short*)(ws + 0);        // 2 MB
    unsigned short* zlo = (unsigned short*)(ws + 2097152);  // 2 MB
    float* norm_n  = (float*)(ws + 4194304);                // 32 KB
    float* norm_p  = (float*)(ws + 4227072);                // 32 KB
    float* colpart = (float*)(ws + 4259840);                // 64 KB
    float* partial = (float*)(ws + 4325376);                // 2 MB
    float* den     = (float*)(ws + 6422528);                // 32 KB

    normalize_split<<<2048, 256, 0, stream>>>(nodes, pair, zhi, zlo, norm_n, norm_p);
    colsum_part<<<dim3(64, 2), 128, 0, stream>>>(nodes, pair, norm_n, norm_p, colpart);
    sim_exp_rowsum<<<dim3(64, 64), 256, 0, stream>>>(zhi, zlo, partial);
    den_reduce<<<32, 256, 0, stream>>>(partial, den);
    finalize<<<1, 256, 0, stream>>>(colpart, den, out);
}
